// Encoder_1408749273269
// MI455X (gfx1250) — matrix operand order repaired and checked by reading, not compiled, not run on hardware
//
#include <hip/hip_runtime.h>

// ---------------------------------------------------------------- constants
#define NL     4      // layers
#define HID    512    // hidden
#define NIN    128    // input features
#define TS     1024   // timesteps
#define BB     32     // batch
#define LH     2048   // NL*HID  (feedback concat width)
#define G4     2048   // 4*HID   (gate width per layer)
#define NWG    128    // persistent workgroups
#define TPB    256    // threads per block (8 waves)

typedef unsigned short ushort_t;
typedef __attribute__((ext_vector_type(16))) __bf16        bf16x16;
typedef __attribute__((ext_vector_type(8)))  float         f32x8;
typedef __attribute__((ext_vector_type(4)))  unsigned int  u32x4;

union Pack32 { u32x4 q[2]; bf16x16 v; };

// ---------------------------------------------------------------- helpers
static __device__ __forceinline__ ushort_t f2bf(float f) {
  unsigned u = __float_as_uint(f);
  unsigned r = (u + 0x7FFFu + ((u >> 16) & 1u)) >> 16;   // round-to-nearest-even
  return (ushort_t)r;
}
static __device__ __forceinline__ float bf2f(ushort_t h) {
  return __uint_as_float(((unsigned)h) << 16);
}
static __device__ __forceinline__ float sigm(float x) {
  return 1.0f / (1.0f + __expf(-x));
}

// D = A[M=16 rows @ m0, K] @ B[N=16 rows @ n0, K]^T + acc, bf16 in / f32 out.
// A row-major [*, lda], Bw row-major [*, ldb] (i.e. weight rows are the N dim).
// Per documented 16-bit WMMA layouts, each lane needs two contiguous 16B loads
// per operand -> straight global_load_b128 from L2-resident weights.
static __device__ __forceinline__ f32x8
wmma_tile(const ushort_t* __restrict__ A, int lda, int m0,
          const ushort_t* __restrict__ Bw, int ldb, int n0,
          int K, f32x8 acc, int lane) {
  const int half = lane >> 4;
  const ushort_t* arow = A  + (size_t)(m0 + (lane & 15)) * lda + half * 8;
  const ushort_t* brow = Bw + (size_t)(n0 + (lane & 15)) * ldb + half * 8;
#pragma unroll 4
  for (int k = 0; k < K; k += 32) {
    Pack32 a, b;
    a.q[0] = *(const u32x4*)(arow + k);        // K = half*8   .. +7
    a.q[1] = *(const u32x4*)(arow + k + 16);   // K = 16+half*8.. +7
    b.q[0] = *(const u32x4*)(brow + k);        // K = half*8   .. +7
    b.q[1] = *(const u32x4*)(brow + k + 16);   // K = 16+half*8.. +7
    __builtin_prefetch(brow + k + 128, 0, 1);  // global_prefetch_b8 of next chunk
    acc = __builtin_amdgcn_wmma_f32_16x16x32_bf16(
        false, a.v, false, b.v, (short)0, acc, false, false);
  }
  return acc;
}

// C/D 16x16 f32 tile <-> memory (row-major [*, ldo]); VGPR r holds
// (M = r + 8*(lane>=16), N = lane&15).
static __device__ __forceinline__ void
store_tile(float* base, int ldo, int m0, int n0, f32x8 c, int lane) {
  int n = lane & 15, mh = (lane >> 4) * 8;
  float* p = base + (size_t)(m0 + mh) * ldo + n0 + n;
#pragma unroll
  for (int r = 0; r < 8; r++) p[(size_t)r * ldo] = c[r];
}
static __device__ __forceinline__ f32x8
load_tile(const float* base, int ldo, int m0, int n0, int lane) {
  int n = lane & 15, mh = (lane >> 4) * 8;
  const float* p = base + (size_t)(m0 + mh) * ldo + n0 + n;
  f32x8 c;
#pragma unroll
  for (int r = 0; r < 8; r++) c[r] = p[(size_t)r * ldo];
  return c;
}

// Device-wide generation barrier (all NWG workgroups are co-resident).
static __device__ __forceinline__ void gbarrier(unsigned* cnt, unsigned target) {
  __syncthreads();
  if (threadIdx.x == 0) {
    __threadfence();              // release stores
    atomicAdd(cnt, 1u);
    while (atomicAdd(cnt, 0u) < target) __builtin_amdgcn_s_sleep(2);
    __threadfence();              // acquire
  }
  __syncthreads();
}

// ---------------------------------------------------------------- pre-pass
__global__ __launch_bounds__(256) void cvt_kernel(const float* __restrict__ s,
                                                  ushort_t* __restrict__ d, int n) {
  int stride = gridDim.x * blockDim.x;
  for (int i = blockIdx.x * blockDim.x + threadIdx.x; i < n; i += stride)
    d[i] = f2bf(s[i]);
}

__global__ void zero_kernel(unsigned* p) { *p = 0u; }

// xp = x @ lin_w^T + lin_b : [T*B=32768, 128] x [512, 128]^T -> bf16 [T*B, 512]
__global__ __launch_bounds__(256) void xp_gemm_kernel(
    const ushort_t* __restrict__ xbf, const ushort_t* __restrict__ lwbf,
    const float* __restrict__ lin_b, ushort_t* __restrict__ xpbf) {
  int gtid = blockIdx.x * blockDim.x + threadIdx.x;
  int lane = gtid & 31;
  int gwid = gtid >> 5;
  int nw = (gridDim.x * blockDim.x) >> 5;
  const int MT = (TS * BB) / 16, NT = HID / 16;            // 2048 x 32 tiles
  for (int tile = gwid; tile < MT * NT; tile += nw) {
    int m0 = (tile >> 5) * 16;
    int n0 = (tile & 31) * 16;
    f32x8 acc = {0, 0, 0, 0, 0, 0, 0, 0};
    acc = wmma_tile(xbf, NIN, m0, lwbf, NIN, n0, NIN, acc, lane);
    int n = lane & 15, mh = (lane >> 4) * 8;
    float bias = lin_b[n0 + n];
    ushort_t* p = xpbf + (size_t)(m0 + mh) * HID + n0 + n;
#pragma unroll
    for (int r = 0; r < 8; r++) p[(size_t)r * HID] = f2bf(acc[r] + bias);
  }
}

// ---------------------------------------------------------------- persistent RNN
__global__ __launch_bounds__(TPB) void lstm_persistent(
    const ushort_t* __restrict__ xpbf,   // [T,B,H] bf16 (input projection)
    const ushort_t* __restrict__ Wbf,    // [L,4H,H] bf16
    const ushort_t* __restrict__ Ubf,    // [L,4H,LH] bf16
    const float* __restrict__ Gp,        // [L,H]
    const float* __restrict__ mask_u,    // [L,B,H]
    const float* __restrict__ mask_w,    // [L,B,H]
    const float* __restrict__ h0, const float* __restrict__ c0,
    float* __restrict__ hstate, float* __restrict__ cstate,   // [L,B,H]
    ushort_t* __restrict__ hxc,          // [B,LH] bf16  gated-hidden concat
    ushort_t* __restrict__ Abuf,         // [B,H]  bf16  current layer input
    float* __restrict__ Upart,           // [L,B,4H]
    float* __restrict__ gbuf,            // [B,4H]
    float* __restrict__ out,             // hT | cT | gates[L,B,T]
    unsigned* __restrict__ barc) {
  const int tid  = threadIdx.x;
  const int gtid = blockIdx.x * blockDim.x + tid;
  const int nthr = gridDim.x * blockDim.x;          // 32768
  const int lane = tid & 31;
  const int gwid = gtid >> 5;                       // 0..1023
  const int GOFF = 2 * NL * BB * HID;               // gates output offset
  unsigned bar = 0;

  // init state from h0/c0 (deterministic per call)
  for (int i = gtid; i < NL * BB * HID; i += nthr) {
    hstate[i] = h0[i];
    cstate[i] = c0[i];
  }
  bar += gridDim.x; gbarrier(barc, bar);

  for (int t = 0; t < TS; ++t) {
    // ---- Phase A: gh gate + gated-hidden concat + layer-0 input ----
    if (gwid < NL * BB) {                            // 128 waves, 1 per (l,b)
      int l = gwid >> 5, b = gwid & 31;
      const float* hrow = hstate + (size_t)(l * BB + b) * HID;
      const float* mrow = mask_u + (size_t)(l * BB + b) * HID;
      const float* grow = Gp + (size_t)l * HID;
      float hm[16], partial = 0.0f;
#pragma unroll
      for (int j = 0; j < 16; j++) {
        int idx = j * 32 + lane;
        float v = hrow[idx] * mrow[idx];
        hm[j] = v;
        partial += v * grow[idx];
      }
#pragma unroll
      for (int off = 16; off; off >>= 1)
        partial += __shfl_xor(partial, off, 32);
      float gh = sigm(partial);
      ushort_t* dst = hxc + (size_t)b * LH + (size_t)l * HID;
#pragma unroll
      for (int j = 0; j < 16; j++) dst[j * 32 + lane] = f2bf(gh * hm[j]);
      if (lane == 0)
        out[GOFF + (size_t)(l * BB + b) * TS + t] = gh;
    }
    if (gtid < BB * HID) {                           // layer-0 masked input
      float xv = bf2f(xpbf[(size_t)t * BB * HID + gtid]);
      Abuf[gtid] = f2bf(xv * mask_w[gtid]);
    }
    bar += gridDim.x; gbarrier(barc, bar);

    // ---- Phase B: Upart[l] = hx_cat @ U[l]^T, all layers in parallel ----
    {                                                // 1024 tiles, 1 per wave
      int l  = gwid >> 8;
      int id = gwid & 255;
      int m0 = (id >> 7) * 16;
      int n0 = (id & 127) * 16;
      f32x8 acc = {0, 0, 0, 0, 0, 0, 0, 0};
      acc = wmma_tile(hxc, LH, m0, Ubf + (size_t)l * G4 * LH, LH, n0, LH, acc, lane);
      store_tile(Upart + (size_t)l * BB * G4, G4, m0, n0, acc, lane);
    }
    bar += gridDim.x; gbarrier(barc, bar);

    // ---- Phase C: layers (sequential) ----
    for (int l = 0; l < NL; ++l) {
      // C2: gates = Abuf @ W[l]^T + Upart[l]
      if (gwid < 256) {
        int m0 = (gwid >> 7) * 16;
        int n0 = (gwid & 127) * 16;
        f32x8 acc = load_tile(Upart + (size_t)l * BB * G4, G4, m0, n0, lane);
        acc = wmma_tile(Abuf, HID, m0, Wbf + (size_t)l * G4 * HID, HID, n0, HID,
                        acc, lane);
        store_tile(gbuf, G4, m0, n0, acc, lane);
      }
      bar += gridDim.x; gbarrier(barc, bar);

      // C3: LSTM cell update + produce next layer's masked bf16 input
      if (gtid < BB * HID) {
        int b = gtid >> 9, h = gtid & 511;
        const float* gr = gbuf + (size_t)b * G4;
        float ii = gr[h], ff = gr[512 + h], gg = gr[1024 + h], oo = gr[1536 + h];
        int si = (l * BB + b) * HID + h;
        float cy = sigm(ff) * cstate[si] + sigm(ii) * tanhf(gg);
        float hy = sigm(oo) * tanhf(cy);
        cstate[si] = cy;
        hstate[si] = hy;
        if (l < NL - 1)
          Abuf[gtid] = f2bf(hy * mask_w[(size_t)(l + 1) * BB * HID + gtid]);
      }
      bar += gridDim.x; gbarrier(barc, bar);
    }
  }

  // ---- final outputs: hT, cT ----
  for (int i = gtid; i < NL * BB * HID; i += nthr) {
    out[i] = hstate[i];
    out[NL * BB * HID + i] = cstate[i];
  }
}

// ---------------------------------------------------------------- launch
extern "C" void kernel_launch(void* const* d_in, const int* in_sizes, int n_in,
                              void* d_out, int out_size, void* d_ws, size_t ws_size,
                              hipStream_t stream) {
  (void)in_sizes; (void)n_in; (void)out_size;
  const float* x      = (const float*)d_in[0];
  const float* lin_w  = (const float*)d_in[1];
  const float* lin_b  = (const float*)d_in[2];
  const float* W      = (const float*)d_in[3];
  const float* U      = (const float*)d_in[4];
  const float* G      = (const float*)d_in[5];
  const float* mask_u = (const float*)d_in[6];
  const float* mask_w = (const float*)d_in[7];
  const float* h0     = (const float*)d_in[8];
  const float* c0     = (const float*)d_in[9];
  float* out = (float*)d_out;
  char* ws = (char*)d_ws;

  size_t off = 0;
  auto alloc = [&](size_t b) { size_t o = off; off += (b + 255) & ~(size_t)255; return o; };
  ushort_t* Wbf   = (ushort_t*)(ws + alloc((size_t)NL * G4 * HID * 2));   //  8.4 MB
  ushort_t* Ubf   = (ushort_t*)(ws + alloc((size_t)NL * G4 * LH * 2));    // 33.6 MB
  ushort_t* lwbf  = (ushort_t*)(ws + alloc((size_t)HID * NIN * 2));
  ushort_t* xbf   = (ushort_t*)(ws + alloc((size_t)TS * BB * NIN * 2));   //  8.4 MB
  ushort_t* xpbf  = (ushort_t*)(ws + alloc((size_t)TS * BB * HID * 2));   // 33.6 MB
  float*    hst   = (float*)(ws + alloc((size_t)NL * BB * HID * 4));
  float*    cst   = (float*)(ws + alloc((size_t)NL * BB * HID * 4));
  ushort_t* hxc   = (ushort_t*)(ws + alloc((size_t)BB * LH * 2));
  ushort_t* Abuf  = (ushort_t*)(ws + alloc((size_t)BB * HID * 2));
  float*    Upart = (float*)(ws + alloc((size_t)NL * BB * G4 * 4));
  float*    gbuf  = (float*)(ws + alloc((size_t)BB * G4 * 4));
  unsigned* barc  = (unsigned*)(ws + alloc(256));
  if (off > ws_size) return;   // workspace too small: refuse to scribble OOB

  cvt_kernel<<<512, 256, 0, stream>>>(W, Wbf, NL * G4 * HID);
  cvt_kernel<<<2048, 256, 0, stream>>>(U, Ubf, NL * G4 * LH);
  cvt_kernel<<<64, 256, 0, stream>>>(lin_w, lwbf, HID * NIN);
  cvt_kernel<<<512, 256, 0, stream>>>(x, xbf, TS * BB * NIN);
  zero_kernel<<<1, 1, 0, stream>>>(barc);
  xp_gemm_kernel<<<256, 256, 0, stream>>>(xbf, lwbf, lin_b, xpbf);
  lstm_persistent<<<NWG, TPB, 0, stream>>>(xpbf, Wbf, Ubf, G, mask_u, mask_w,
                                           h0, c0, hst, cst, hxc, Abuf, Upart,
                                           gbuf, out, barc);
}
